// WorldModel_19378892439626
// MI455X (gfx1250) — matrix operand order repaired and checked by reading, not compiled, not run on hardware
//
#include <hip/hip_runtime.h>

#define CAT 2048
#define NODE 1024
#define HOR 8
#define LOG2PI_F 1.8378770664093453f

typedef __attribute__((ext_vector_type(16))) __bf16 bf16x16;
typedef __attribute__((ext_vector_type(8)))  float  f32x8;

union FragU { bf16x16 v; uint4 q[2]; };

__device__ __forceinline__ unsigned short f32_to_bf16(float f) {
  unsigned int u = __float_as_uint(f);
  u += 0x7FFFu + ((u >> 16) & 1u);           // round-to-nearest-even
  return (unsigned short)(u >> 16);
}

// ---------------- init ----------------
__global__ void k_init(float* loss) { if (threadIdx.x == 0) *loss = 0.0f; }

// ---------------- obs_mean[c] = relu(Wo1[c,:]+bo1)·Wo2 + bo2 ----------------
__global__ void k_obs_mean(const float* __restrict__ Wo1, const float* __restrict__ bo1,
                           const float* __restrict__ Wo2, const float* __restrict__ bo2,
                           float* __restrict__ om) {
  __shared__ float red[256];
  const int c = blockIdx.x, tid = threadIdx.x;
  float acc = 0.0f;
  for (int n = tid; n < NODE; n += 256)
    acc += fmaxf(Wo1[(size_t)c * NODE + n] + bo1[n], 0.0f) * Wo2[n];
  red[tid] = acc; __syncthreads();
  for (int s = 128; s > 0; s >>= 1) { if (tid < s) red[tid] += red[tid + s]; __syncthreads(); }
  if (tid == 0) om[c] = red[0] + bo2[0];
}

// ---------------- Wt2 (K=1024 x N=2048, f32) -> bf16, transposed to [n][k] ----------------
__global__ void k_wt2t(const float* __restrict__ Wt2, unsigned short* __restrict__ out) {
  const int id = blockIdx.x * 256 + threadIdx.x;   // 2048*1024 threads
  const int n = id >> 10, k = id & 1023;
  out[id] = f32_to_bf16(Wt2[(size_t)k * CAT + n]);
}

// ---------------- hidden_t[c,n] = relu(Wt1[c,n] + a*Wt1[C,n] + bt1[n]) as bf16 -----------
__global__ void k_hidden_t(const float* __restrict__ Wt1, const float* __restrict__ bt1,
                           const float* __restrict__ actions, int t,
                           unsigned short* __restrict__ ht) {
  const int id = blockIdx.x * 256 + threadIdx.x;   // 2048*1024 threads, id = c*1024+n
  const int n = id & 1023;
  const float a = actions[t];
  float v = Wt1[id] + a * Wt1[(size_t)CAT * NODE + n] + bt1[n];
  ht[id] = f32_to_bf16(fmaxf(v, 0.0f));
}

// ---------------- WMMA GEMM: logits = hidden_t(2048x1024) @ Wt2(1024x2048) + bt2 ---------
// A: bf16 row-major [M=2048][K=1024]; Bt: bf16 [N=2048][K=1024] (K contiguous).
// Per wave: 64x64 output tile = 4x4 WMMA accumulators (16 WMMAs per 16 b128 loads / K-step).
__global__ __launch_bounds__(256) void k_gemm_wmma(
    const unsigned short* __restrict__ A, const unsigned short* __restrict__ Bt,
    const float* __restrict__ bias, float* __restrict__ Cmat) {
  const int wave = blockIdx.x * 8 + (threadIdx.x >> 5);  // 1024 waves total
  const int lane = threadIdx.x & 31;
  const int l16 = lane & 15;
  const int half = lane >> 4;
  const int tm = wave >> 5;   // 0..31  (M tiles of 64)
  const int tn = wave & 31;   // 0..31  (N tiles of 64)
  const int m0 = tm * 64;
  const int n0 = tn * 64;

  f32x8 acc[4][4];
  const f32x8 zero = {0.f, 0.f, 0.f, 0.f, 0.f, 0.f, 0.f, 0.f};
#pragma unroll
  for (int im = 0; im < 4; ++im)
#pragma unroll
    for (int in = 0; in < 4; ++in) acc[im][in] = zero;

#pragma unroll 1
  for (int kb = 0; kb < NODE; kb += 32) {
    FragU a[4], b[4];
    // A fragment (16-bit A layout): elem e<8 -> K = kb+8*half+e ; e>=8 -> K = kb+16+8*half+(e-8)
    // B fragment (16-bit B layout): lane n = l16, elem e<8 -> K = kb+8*half+e ; e>=8 -> K = kb+16+8*half+(e-8)
#pragma unroll
    for (int i = 0; i < 4; ++i) {
      const unsigned short* pa = A + (size_t)(m0 + i * 16 + l16) * NODE + kb + half * 8;
      a[i].q[0] = *(const uint4*)(pa);
      a[i].q[1] = *(const uint4*)(pa + 16);
      const unsigned short* pb = Bt + (size_t)(n0 + i * 16 + l16) * NODE + kb + half * 8;
      b[i].q[0] = *(const uint4*)(pb);
      b[i].q[1] = *(const uint4*)(pb + 16);
    }
#pragma unroll
    for (int im = 0; im < 4; ++im)
#pragma unroll
      for (int in = 0; in < 4; ++in)
        acc[im][in] = __builtin_amdgcn_wmma_f32_16x16x32_bf16(
            false, a[im].v, false, b[in].v, (short)0, acc[im][in], false, false);
  }

  // Epilogue: C/D layout -> n = n0+in*16+l16, m = m0+im*16+8*half+r
#pragma unroll
  for (int im = 0; im < 4; ++im) {
#pragma unroll
    for (int in = 0; in < 4; ++in) {
      const int n = n0 + in * 16 + l16;
      const float bv = bias[n];
      const int mb = m0 + im * 16 + half * 8;
#pragma unroll
      for (int r = 0; r < 8; ++r)
        Cmat[(size_t)(mb + r) * CAT + n] = acc[im][in][r] + bv;
    }
  }
}

// ---------------- per-row softmax stats of logits + prev scaling ----------------
__global__ void k_rowstats(const float* __restrict__ logits, const float* __restrict__ prev,
                           float* __restrict__ rowmax, float* __restrict__ rowscale) {
  __shared__ float red[256];
  const int c = blockIdx.x, tid = threadIdx.x;
  const float* row = logits + (size_t)c * CAT;
  float m = -3.402823466e+38f;
  for (int j = tid; j < CAT; j += 256) m = fmaxf(m, row[j]);
  red[tid] = m; __syncthreads();
  for (int s = 128; s > 0; s >>= 1) { if (tid < s) red[tid] = fmaxf(red[tid], red[tid + s]); __syncthreads(); }
  m = red[0]; __syncthreads();
  float sum = 0.0f;
  for (int j = tid; j < CAT; j += 256) sum += __expf(row[j] - m);
  red[tid] = sum; __syncthreads();
  for (int s = 128; s > 0; s >>= 1) { if (tid < s) red[tid] += red[tid + s]; __syncthreads(); }
  if (tid == 0) { rowmax[c] = m; rowscale[c] = prev[c] / red[0]; }
}

// ---------------- pred[j] = sum_c rowscale[c]*exp(logits[c][j]-rowmax[c]) ----------------
__global__ void k_pred(const float* __restrict__ logits, const float* __restrict__ rowmax,
                       const float* __restrict__ rowscale, float* __restrict__ pred) {
  const int j = blockIdx.x * 256 + threadIdx.x;   // 2048 threads
  float acc = 0.0f;
  for (int c = 0; c < CAT; ++c)
    acc += rowscale[c] * __expf(logits[(size_t)c * CAT + j] - rowmax[c]);
  pred[j] = acc;
}

// ---------------- belief MLP layer 1 ----------------
__global__ void k_hidden_b(const float* __restrict__ Wb1, const float* __restrict__ bb1,
                           const float* __restrict__ prev, const float* __restrict__ actions,
                           const float* __restrict__ obs, int t, float* __restrict__ hb) {
  const int n = blockIdx.x * 256 + threadIdx.x;   // 1024 threads
  float acc = bb1[n];
  for (int i = 0; i < CAT; ++i) acc += prev[i] * Wb1[(size_t)i * NODE + n];
  acc += actions[t] * Wb1[(size_t)CAT * NODE + n];
  acc += obs[t]     * Wb1[(size_t)(CAT + 1) * NODE + n];
  hb[n] = fmaxf(acc, 0.0f);
}

// ---------------- belief MLP layer 2 ----------------
__global__ void k_logits_b(const float* __restrict__ hb, const float* __restrict__ Wb2,
                           const float* __restrict__ bb2, float* __restrict__ lb) {
  const int j = blockIdx.x * 256 + threadIdx.x;   // 2048 threads
  float acc = bb2[j];
  for (int n = 0; n < NODE; ++n) acc += hb[n] * Wb2[(size_t)n * CAT + j];
  lb[j] = acc;
}

// ---------------- belief = softmax(logits_b); loss += first term ----------------
__global__ void k_belief_first(const float* __restrict__ lb, const float* __restrict__ om,
                               const float* __restrict__ obs, int t,
                               float* __restrict__ belief, float* __restrict__ loss) {
  __shared__ float red[256];
  const int tid = threadIdx.x;
  float m = -3.402823466e+38f;
  for (int j = tid; j < CAT; j += 256) m = fmaxf(m, lb[j]);
  red[tid] = m; __syncthreads();
  for (int s = 128; s > 0; s >>= 1) { if (tid < s) red[tid] = fmaxf(red[tid], red[tid + s]); __syncthreads(); }
  m = red[0]; __syncthreads();
  float sum = 0.0f;
  for (int j = tid; j < CAT; j += 256) sum += __expf(lb[j] - m);
  red[tid] = sum; __syncthreads();
  for (int s = 128; s > 0; s >>= 1) { if (tid < s) red[tid] += red[tid + s]; __syncthreads(); }
  const float inv = 1.0f / red[0]; __syncthreads();
  const float o = obs[t];
  float acc = 0.0f;
  for (int j = tid; j < CAT; j += 256) {
    const float b = __expf(lb[j] - m) * inv;
    belief[j] = b;
    const float d = o - om[j];
    acc += b * (0.5f * d * d + 0.5f * LOG2PI_F);   // = -b*logp
  }
  red[tid] = acc; __syncthreads();
  for (int s = 128; s > 0; s >>= 1) { if (tid < s) red[tid] += red[tid + s]; __syncthreads(); }
  if (tid == 0) *loss += red[0];
}

// ---------------- loss += KL(belief || pred) ----------------
__global__ void k_second(const float* __restrict__ belief, const float* __restrict__ pred,
                         float* __restrict__ loss) {
  __shared__ float red[256];
  const int tid = threadIdx.x;
  float acc = 0.0f;
  for (int j = tid; j < CAT; j += 256)
    acc += belief[j] * (__logf(belief[j]) - __logf(pred[j]));
  red[tid] = acc; __syncthreads();
  for (int s = 128; s > 0; s >>= 1) { if (tid < s) red[tid] += red[tid + s]; __syncthreads(); }
  if (tid == 0) *loss += red[0];
}

// ---------------- write outputs ----------------
__global__ void k_finalize(const float* __restrict__ loss, const float* __restrict__ belief,
                           float* __restrict__ out) {
  const int i = blockIdx.x * 256 + threadIdx.x;
  if (i == 0) out[0] = *loss;
  if (i < CAT) out[1 + i] = belief[i];
}

extern "C" void kernel_launch(void* const* d_in, const int* in_sizes, int n_in,
                              void* d_out, int out_size, void* d_ws, size_t ws_size,
                              hipStream_t stream) {
  const float* obs     = (const float*)d_in[0];
  const float* actions = (const float*)d_in[1];
  const float* prev0   = (const float*)d_in[2];
  const float* Wb1 = (const float*)d_in[3];  const float* bb1 = (const float*)d_in[4];
  const float* Wb2 = (const float*)d_in[5];  const float* bb2 = (const float*)d_in[6];
  const float* Wt1 = (const float*)d_in[7];  const float* bt1 = (const float*)d_in[8];
  const float* Wt2 = (const float*)d_in[9];  const float* bt2 = (const float*)d_in[10];
  const float* Wo1 = (const float*)d_in[11]; const float* bo1 = (const float*)d_in[12];
  const float* Wo2 = (const float*)d_in[13]; const float* bo2 = (const float*)d_in[14];

  char* ws = (char*)d_ws;
  size_t off = 0;
  float*          logits  = (float*)(ws + off);          off += (size_t)CAT * CAT * 4;   // 16.78 MB
  unsigned short* hiddenT = (unsigned short*)(ws + off); off += (size_t)CAT * NODE * 2;  // 4.19 MB
  unsigned short* wt2t    = (unsigned short*)(ws + off); off += (size_t)CAT * NODE * 2;  // 4.19 MB
  float* obs_mean = (float*)(ws + off); off += CAT * 4;
  float* hidden_b = (float*)(ws + off); off += NODE * 4;
  float* logits_b = (float*)(ws + off); off += CAT * 4;
  float* belief0  = (float*)(ws + off); off += CAT * 4;
  float* belief1  = (float*)(ws + off); off += CAT * 4;
  float* rowmax   = (float*)(ws + off); off += CAT * 4;
  float* rowscale = (float*)(ws + off); off += CAT * 4;
  float* pred     = (float*)(ws + off); off += CAT * 4;
  float* loss     = (float*)(ws + off); off += 256;

  k_init<<<1, 64, 0, stream>>>(loss);
  k_obs_mean<<<CAT, 256, 0, stream>>>(Wo1, bo1, Wo2, bo2, obs_mean);
  k_wt2t<<<(CAT * NODE) / 256, 256, 0, stream>>>(Wt2, wt2t);

  const float* prev = prev0;
  float* bufs[2] = {belief0, belief1};
  for (int t = 0; t < HOR; ++t) {
    float* cur = bufs[t & 1];
    k_hidden_t<<<(CAT * NODE) / 256, 256, 0, stream>>>(Wt1, bt1, actions, t, hiddenT);
    k_gemm_wmma<<<128, 256, 0, stream>>>(hiddenT, wt2t, bt2, logits);
    k_rowstats<<<CAT, 256, 0, stream>>>(logits, prev, rowmax, rowscale);
    k_pred<<<CAT / 256, 256, 0, stream>>>(logits, rowmax, rowscale, pred);
    k_hidden_b<<<NODE / 256, 256, 0, stream>>>(Wb1, bb1, prev, actions, obs, t, hidden_b);
    k_logits_b<<<CAT / 256, 256, 0, stream>>>(hidden_b, Wb2, bb2, logits_b);
    k_belief_first<<<1, 256, 0, stream>>>(logits_b, obs_mean, obs, t, cur, loss);
    k_second<<<1, 256, 0, stream>>>(cur, pred, loss);
    prev = cur;
  }
  k_finalize<<<(CAT + 256) / 256 + 1, 256, 0, stream>>>(loss, prev, (float*)d_out);
}
